// GCN_58514634441241
// MI455X (gfx1250) — matrix operand order repaired and checked by reading, not compiled, not run on hardware
//
#include <hip/hip_runtime.h>
#include <hip/hip_bf16.h>
#include <stdint.h>

#define N_NODES   100000
#define N_EDGES   1600000
#define CH        128
#define NUM_GRAPHS 512
#define OUT_CH    2
#define N_TILES   (N_NODES / 16)   // 6250 row-tiles of 16 nodes

typedef __bf16 bf16;
typedef __attribute__((ext_vector_type(16))) __bf16 v16bf;
typedef __attribute__((ext_vector_type(8)))  float  v8f;

// round-to-nearest-even f32 -> bf16 via integer ops (no dependence on cvt lowering)
__device__ __forceinline__ bf16 f2bf(float f) {
    uint32_t u = __builtin_bit_cast(uint32_t, f);
    uint32_t r = (u + 0x7FFFu + ((u >> 16) & 1u)) >> 16;
    return __builtin_bit_cast(bf16, (uint16_t)r);
}

// ---------------------------------------------------------------------------
// Pack a 128x128 f32 weight matrix w[k][n] into the WMMA A-operand layout for
// A = w^T tiles (M=out_ch, K=in_ch), bf16. Linear order:
//   o = ((mt*4 + kb)*32 + lane)*16 + e      (mt: 8 M-tiles, kb: 4 K-blocks)
// k order packed here (same as the B operand): lane half h, elem e:
//   K = kb*32 + h*16 + e (contiguous within each 32-wide K-block)
// ---------------------------------------------------------------------------
__global__ void fmt_weights(const float* __restrict__ w, bf16* __restrict__ out) {
    int o = blockIdx.x * blockDim.x + threadIdx.x;
    if (o >= CH * CH) return;
    int e    = o & 15;
    int lane = (o >> 4) & 31;
    int kb   = (o >> 9) & 3;
    int mt   = o >> 11;
    int half = lane >> 4;
    int M = mt * 16 + (lane & 15);                       // out channel
    int ke = half * 16 + e;
    int K = kb * 32 + ke;                                // in channel
    out[o] = f2bf(w[K * CH + M]);                        // w^T[M][K] = w[K][M]
}

// ---------------------------------------------------------------------------
// Edge scatter-add: agg[dst] += x[src].  One wave per edge, float4 per lane.
// ---------------------------------------------------------------------------
__global__ void scatter_add(const float* __restrict__ x, const int* __restrict__ src,
                            const int* __restrict__ dst, float* __restrict__ agg) {
    int t = blockIdx.x * blockDim.x + threadIdx.x;
    int e = t >> 5;
    int lane = t & 31;
    if (e >= N_EDGES) return;
    int s = src[e], d = dst[e];
    float4 v = ((const float4*)(x + (size_t)s * CH))[lane];
    float* ap = agg + (size_t)d * CH + lane * 4;
    atomicAdd(ap + 0, v.x);
    atomicAdd(ap + 1, v.y);
    atomicAdd(ap + 2, v.z);
    atomicAdd(ap + 3, v.w);
}

// ---------------------------------------------------------------------------
// Fused GIN MLP: out = [relu]( relu((xin+agg) @ w1 + b1) @ w2 + b2 )
// Computed transposed: h^T = w^T · x^T, one wave per 16-node tile.
// D tile layout: lane L -> node col L%16; elem r -> channel r + 8*(L/16).
// `out` may alias `agg` (per-node disjoint read-before-write within a wave).
// ---------------------------------------------------------------------------
__global__ void __launch_bounds__(256)
gin_mlp(const float* __restrict__ xin, const float* agg, float* out,
        const v16bf* __restrict__ w1a, const float* __restrict__ b1,
        const v16bf* __restrict__ w2a, const float* __restrict__ b2,
        int final_relu) {
    int wid  = (blockIdx.x * blockDim.x + threadIdx.x) >> 5;
    int lane = threadIdx.x & 31;
    if (wid >= N_TILES) return;
    int half = lane >> 4;
    int node = wid * 16 + (lane & 15);
    const float* xr = xin + (size_t)node * CH;
    const float* ar = agg + (size_t)node * CH;

    // B operand (x^T tiles): lane -> node, elem i -> ch = kb*32 + half*16 + i
    v16bf bx[4];
#pragma unroll
    for (int kb = 0; kb < 4; ++kb) {
        int ch0 = kb * 32 + half * 16;
        const float4* xv = (const float4*)(xr + ch0);
        const float4* av = (const float4*)(ar + ch0);
#pragma unroll
        for (int q = 0; q < 4; ++q) {
            float4 a = xv[q];
            float4 b = av[q];
            bx[kb][q * 4 + 0] = f2bf(a.x + b.x);
            bx[kb][q * 4 + 1] = f2bf(a.y + b.y);
            bx[kb][q * 4 + 2] = f2bf(a.z + b.z);
            bx[kb][q * 4 + 3] = f2bf(a.w + b.w);
        }
    }

    // GEMM1: d[mt] = relu(w1^T(tile mt) · x^T + b1)
    v8f d[8];
#pragma unroll
    for (int mt = 0; mt < 8; ++mt) {
        v8f acc;
#pragma unroll
        for (int r = 0; r < 8; ++r) acc[r] = b1[mt * 16 + half * 8 + r];
#pragma unroll
        for (int kb = 0; kb < 4; ++kb) {
            v16bf wa = w1a[(mt * 4 + kb) * 32 + lane];
            acc = __builtin_amdgcn_wmma_f32_16x16x32_bf16(
                false, wa, false, bx[kb], (short)0, acc, false, false);
        }
#pragma unroll
        for (int r = 0; r < 8; ++r) acc[r] = fmaxf(acc[r], 0.0f);
        d[mt] = acc;
    }

    // Re-pack D tiles into B operand for GEMM2 via half-wave exchange:
    // B elem i must hold h_mid[ch = kb*32 + half*16 + i][node].
    v16bf bh[4];
#pragma unroll
    for (int kb = 0; kb < 4; ++kb) {
#pragma unroll
        for (int i = 0; i < 8; ++i) {
            float d0 = d[kb * 2][i];
            float d1 = d[kb * 2 + 1][i];
            float s0 = __shfl_xor(d0, 16, 32);
            float s1 = __shfl_xor(d1, 16, 32);
            bh[kb][i]     = f2bf(half ? s1 : d0);
            bh[kb][i + 8] = f2bf(half ? d1 : s0);
        }
    }

    // GEMM2 + bias (+ optional relu), store row-major [node][ch]
#pragma unroll
    for (int mt = 0; mt < 8; ++mt) {
        v8f acc;
#pragma unroll
        for (int r = 0; r < 8; ++r) acc[r] = b2[mt * 16 + half * 8 + r];
#pragma unroll
        for (int kb = 0; kb < 4; ++kb) {
            v16bf wa = w2a[(mt * 4 + kb) * 32 + lane];
            acc = __builtin_amdgcn_wmma_f32_16x16x32_bf16(
                false, wa, false, bh[kb], (short)0, acc, false, false);
        }
        float* op = out + (size_t)node * CH + mt * 16 + half * 8;
#pragma unroll
        for (int r = 0; r < 8; ++r) {
            float v = acc[r];
            if (final_relu) v = fmaxf(v, 0.0f);
            op[r] = v;
        }
    }
}

// ---------------------------------------------------------------------------
// Global mean pool, pass 1: per-graph sums + counts via float atomics.
// ---------------------------------------------------------------------------
__global__ void pool_accum(const float* __restrict__ h, const int* __restrict__ batch,
                           float* __restrict__ sums, float* __restrict__ counts) {
    int t = blockIdx.x * blockDim.x + threadIdx.x;
    int n = t >> 5;
    int lane = t & 31;
    if (n >= N_NODES) return;
    int g = batch[n];
    float4 v = ((const float4*)(h + (size_t)n * CH))[lane];
    float* sp = sums + (size_t)g * CH + lane * 4;
    atomicAdd(sp + 0, v.x);
    atomicAdd(sp + 1, v.y);
    atomicAdd(sp + 2, v.z);
    atomicAdd(sp + 3, v.w);
    if (lane == 0) atomicAdd(&counts[g], 1.0f);
}

// ---------------------------------------------------------------------------
// Head: out[g][o] = (sums[g]/max(cnt,1)) . w_lin[:,o] + b_lin[o]
// ---------------------------------------------------------------------------
__global__ void final_linear(const float* __restrict__ sums, const float* __restrict__ counts,
                             const float* __restrict__ wl, const float* __restrict__ bl,
                             float* __restrict__ out) {
    int t = blockIdx.x * blockDim.x + threadIdx.x;
    if (t >= NUM_GRAPHS * OUT_CH) return;
    int g = t / OUT_CH, o = t % OUT_CH;
    float inv = 1.0f / fmaxf(counts[g], 1.0f);
    float acc = bl[o];
#pragma unroll 8
    for (int k = 0; k < CH; ++k) acc += sums[(size_t)g * CH + k] * inv * wl[k * OUT_CH + o];
    out[t] = acc;
}

extern "C" void kernel_launch(void* const* d_in, const int* in_sizes, int n_in,
                              void* d_out, int out_size, void* d_ws, size_t ws_size,
                              hipStream_t stream) {
    const float* x     = (const float*)d_in[0];
    const int*   ei    = (const int*)d_in[1];
    const int*   src   = ei;
    const int*   dst   = ei + N_EDGES;
    const int*   batch = (const int*)d_in[2];
    const float* w1[3] = {(const float*)d_in[3],  (const float*)d_in[7],  (const float*)d_in[11]};
    const float* b1[3] = {(const float*)d_in[4],  (const float*)d_in[8],  (const float*)d_in[12]};
    const float* w2[3] = {(const float*)d_in[5],  (const float*)d_in[9],  (const float*)d_in[13]};
    const float* b2[3] = {(const float*)d_in[6],  (const float*)d_in[10], (const float*)d_in[14]};
    const float* wl    = (const float*)d_in[15];
    const float* bl    = (const float*)d_in[16];

    const size_t BUF_BYTES = (size_t)N_NODES * CH * sizeof(float);   // 51.2 MB
    char*  ws    = (char*)d_ws;
    float* bufA  = (float*)ws;
    float* bufB  = (float*)(ws + BUF_BYTES);
    bf16*  wfmt  = (bf16*)(ws + 2 * BUF_BYTES);                      // 6 * 16384 bf16
    float* sums  = (float*)(ws + 2 * BUF_BYTES + 6 * (size_t)CH * CH * sizeof(bf16));
    float* cnts  = sums + (size_t)NUM_GRAPHS * CH;

    // Pack all six weight matrices into WMMA A-operand bf16 layout.
    for (int l = 0; l < 3; ++l) {
        fmt_weights<<<(CH * CH + 255) / 256, 256, 0, stream>>>(w1[l], wfmt + (size_t)(2 * l) * CH * CH);
        fmt_weights<<<(CH * CH + 255) / 256, 256, 0, stream>>>(w2[l], wfmt + (size_t)(2 * l + 1) * CH * CH);
    }

    const int scatter_blocks = ((size_t)N_EDGES * 32 + 255) / 256;   // 200000
    const int mlp_blocks     = ((size_t)N_TILES * 32 + 255) / 256;   // 782
    const int pool_blocks    = ((size_t)N_NODES * 32 + 255) / 256;   // 12500

    const float* cur = x;
    for (int l = 0; l < 3; ++l) {
        float* ab = (l == 1) ? bufB : bufA;     // layer0->A, layer1->B, layer2->A
        hipMemsetAsync(ab, 0, BUF_BYTES, stream);
        scatter_add<<<scatter_blocks, 256, 0, stream>>>(cur, src, dst, ab);
        const v16bf* w1a = (const v16bf*)(wfmt + (size_t)(2 * l) * CH * CH);
        const v16bf* w2a = (const v16bf*)(wfmt + (size_t)(2 * l + 1) * CH * CH);
        gin_mlp<<<mlp_blocks, 256, 0, stream>>>(cur, ab, ab, w1a, b1[l], w2a, b2[l],
                                                (l < 2) ? 1 : 0);
        cur = ab;
    }

    hipMemsetAsync(sums, 0, ((size_t)NUM_GRAPHS * CH + NUM_GRAPHS) * sizeof(float), stream);
    pool_accum<<<pool_blocks, 256, 0, stream>>>(cur, batch, sums, cnts);
    final_linear<<<(NUM_GRAPHS * OUT_CH + 255) / 256, 256, 0, stream>>>(
        sums, cnts, wl, bl, (float*)d_out);
}
